// SelfAttention_46093589021317
// MI455X (gfx1250) — matrix operand order repaired and checked by reading, not compiled, not run on hardware
//
#include <hip/hip_runtime.h>

#define BATCH 2
#define SEQ   2048
#define EMB   1024
#define HEADS 16
#define HDIM  64

typedef __attribute__((ext_vector_type(16))) __bf16 v16bf;
typedef __attribute__((ext_vector_type(8)))  float  v8f;

union BF16x16 {
    v16bf v;
    uint4 u4[2];
    unsigned short us[16];
};

// Hardware conversion: lowers to v_cvt_*bf16_f32 instead of a 3-op bit trick.
__device__ __forceinline__ unsigned short f2bf(float f) {
    __bf16 b = (__bf16)f;
    unsigned short u;
    __builtin_memcpy(&u, &b, 2);
    return u;
}

__device__ __forceinline__ v8f wmma_bf16(const BF16x16& A, const BF16x16& B, v8f C) {
    // D = A(16x32 bf16) * B(32x16 bf16) + C(16x16 f32)
    return __builtin_amdgcn_wmma_f32_16x16x32_bf16(
        /*neg_a=*/false, A.v, /*neg_b=*/false, B.v,
        /*c_mod=*/(short)0, C, /*reuse_a=*/false, /*reuse_b=*/false);
}

// ---------------------------------------------------------------------------
// Stage 0: bulk fp32 -> bf16 conversion (used once for W_fc)
// ---------------------------------------------------------------------------
__global__ void __launch_bounds__(256) cvt_kernel(
    const float* __restrict__ src, unsigned short* __restrict__ dst, int n8)
{
    const int i = blockIdx.x * 256 + threadIdx.x;   // 8 elements per thread
    if (i < n8) {
        const float* p = src + (size_t)i * 8;
        float4 f0 = *(const float4*)p;
        float4 f1 = *(const float4*)(p + 4);
        union { uint4 u; unsigned short us[8]; } o;
        o.us[0] = f2bf(f0.x); o.us[1] = f2bf(f0.y);
        o.us[2] = f2bf(f0.z); o.us[3] = f2bf(f0.w);
        o.us[4] = f2bf(f1.x); o.us[5] = f2bf(f1.y);
        o.us[6] = f2bf(f1.z); o.us[7] = f2bf(f1.w);
        *(uint4*)(dst + (size_t)i * 8) = o.u;
    }
}

// ---------------------------------------------------------------------------
// Stage 1: per-head projection  proj[s,e] = sum_d X[s, h*64+d] * W[e,d]
// Output layouts (bf16): transposed==0 -> [b][h][s][d], transposed==1 -> [b][h][d][s]
// ---------------------------------------------------------------------------
__global__ void __launch_bounds__(128) proj_kernel(
    const float* __restrict__ X, const float* __restrict__ W,
    unsigned short* __restrict__ out, int transposed)
{
    const int wave = threadIdx.x >> 5;
    const int lane = threadIdx.x & 31;
    const int b    = blockIdx.x >> 7;         // 2*16*8 = 256 blocks
    const int h    = (blockIdx.x >> 3) & 15;
    const int sgrp = blockIdx.x & 7;

    const int n     = lane & 15;
    const int row   = lane & 15;
    const int koff  = (lane < 16) ? 0 : 8;
    const int bhalf = (lane < 16) ? 0 : 16;
    const int qb    = (lane < 16) ? 0 : 8;

    // Pre-pack W^T as WMMA B-operands: B(d,e) = W[e][d]; 4 e-tiles x 2 K-chunks
    BF16x16 Bw[4][2];
    #pragma unroll
    for (int et = 0; et < 4; ++et)
        #pragma unroll
        for (int c = 0; c < 2; ++c) {
            const float* wp = W + (et*16 + n)*HDIM + c*32 + koff;
            alignas(16) float ff[16];
            *(float4*)&ff[0]  = *(const float4*)(wp + 0);
            *(float4*)&ff[4]  = *(const float4*)(wp + 4);
            *(float4*)&ff[8]  = *(const float4*)(wp + 16);
            *(float4*)&ff[12] = *(const float4*)(wp + 20);
            #pragma unroll
            for (int j = 0; j < 16; ++j) Bw[et][c].us[j] = f2bf(ff[j]);
        }

    const v8f vzero = {0.f,0.f,0.f,0.f,0.f,0.f,0.f,0.f};

    for (int i = 0; i < 4; ++i) {
        const int s0 = (sgrp*16 + wave*4 + i) * 16;
        // A operand: 16 tokens x 64 d (two K=32 chunks), fp32 -> bf16
        BF16x16 A[2];
        #pragma unroll
        for (int c = 0; c < 2; ++c) {
            const float* xp = X + ((size_t)b*SEQ + s0 + row)*EMB + h*HDIM + c*32 + koff;
            alignas(16) float fa[16];
            *(float4*)&fa[0]  = *(const float4*)(xp + 0);
            *(float4*)&fa[4]  = *(const float4*)(xp + 4);
            *(float4*)&fa[8]  = *(const float4*)(xp + 16);
            *(float4*)&fa[12] = *(const float4*)(xp + 20);
            #pragma unroll
            for (int j = 0; j < 16; ++j) A[c].us[j] = f2bf(fa[j]);
        }
        #pragma unroll
        for (int et = 0; et < 4; ++et) {
            v8f acc = vzero;
            acc = wmma_bf16(A[0], Bw[et][0], acc);
            acc = wmma_bf16(A[1], Bw[et][1], acc);
            const int e = et*16 + n;
            #pragma unroll
            for (int r = 0; r < 8; ++r) {
                const int s = s0 + qb + r;
                const unsigned short bv = f2bf(acc[r]);
                if (!transposed)
                    out[(((size_t)b*HEADS + h)*SEQ + s)*HDIM + e] = bv;
                else
                    out[(((size_t)b*HEADS + h)*HDIM + e)*SEQ + s] = bv;
            }
        }
    }
}

// ---------------------------------------------------------------------------
// Stage 2: fused attention with softmax over HEADS.
// Block = 4 waves, one 16-row q-tile; k streamed in tiles of 32.
//   phase1: wave w -> scores for heads 4w..4w+3  -> LDS [q][k][h]
//   phase2: per-(q,k) softmax over 16 heads      -> LDS P [h][q][k] (bf16)
//   phase3: wave w -> out[h][:, d-chunk w] += P_h x V_h   (K=32 WMMA)
// ---------------------------------------------------------------------------
__global__ void __launch_bounds__(128) attn_kernel(
    const unsigned short* __restrict__ Qp, const unsigned short* __restrict__ Kp,
    const unsigned short* __restrict__ Vt, unsigned short* __restrict__ ctx)
{
    __shared__ float          Ssc[16][32][16];  // [q][k][h]  32 KB
    __shared__ unsigned short Pm [16][16][32];  // [h][q][k]  16 KB

    const int wave = threadIdx.x >> 5;
    const int lane = threadIdx.x & 31;
    const int b  = blockIdx.x >> 7;
    const int q0 = (blockIdx.x & 127) * 16;

    const int row   = lane & 15;
    const int n     = lane & 15;
    const int koff  = (lane < 16) ? 0 : 8;
    const int bhalf = (lane < 16) ? 0 : 16;
    const int qb    = (lane < 16) ? 0 : 8;

    const v8f vzero = {0.f,0.f,0.f,0.f,0.f,0.f,0.f,0.f};
    v8f outAcc[HEADS];
    #pragma unroll
    for (int h2 = 0; h2 < HEADS; ++h2) outAcc[h2] = vzero;

    for (int kb = 0; kb < SEQ/32; ++kb) {
        const int kbase = kb*32;

        // ---- phase 1: scores ----
        #pragma unroll
        for (int hh = 0; hh < 4; ++hh) {
            const int h = wave*4 + hh;
            BF16x16 A[2];
            const unsigned short* qp =
                Qp + (((size_t)b*HEADS + h)*SEQ + q0 + row)*HDIM;
            A[0].u4[0] = *(const uint4*)(qp + 0  + koff);
            A[0].u4[1] = *(const uint4*)(qp + 16 + koff);
            A[1].u4[0] = *(const uint4*)(qp + 32 + koff);
            A[1].u4[1] = *(const uint4*)(qp + 48 + koff);
            #pragma unroll
            for (int sub = 0; sub < 2; ++sub) {
                const unsigned short* kp =
                    Kp + (((size_t)b*HEADS + h)*SEQ + kbase + sub*16 + n)*HDIM + koff;
                BF16x16 B0, B1;
                B0.u4[0] = *(const uint4*)(kp + 0);
                B0.u4[1] = *(const uint4*)(kp + 16);
                B1.u4[0] = *(const uint4*)(kp + 32);
                B1.u4[1] = *(const uint4*)(kp + 48);
                v8f acc = vzero;
                acc = wmma_bf16(A[0], B0, acc);
                acc = wmma_bf16(A[1], B1, acc);
                #pragma unroll
                for (int r = 0; r < 8; ++r)
                    Ssc[qb + r][sub*16 + n][h] = acc[r] * 0.03125f; // 1/sqrt(1024)
            }
        }
        __syncthreads();

        // ---- phase 2: softmax over heads (fully in-lane) ----
        #pragma unroll
        for (int ppi = 0; ppi < 4; ++ppi) {
            const int p  = threadIdx.x + ppi*128;   // 0..511 -> (q,k)
            const int q  = p >> 5;
            const int kk = p & 31;
            alignas(16) float sv[16];
            *(float4*)&sv[0]  = *(const float4*)&Ssc[q][kk][0];
            *(float4*)&sv[4]  = *(const float4*)&Ssc[q][kk][4];
            *(float4*)&sv[8]  = *(const float4*)&Ssc[q][kk][8];
            *(float4*)&sv[12] = *(const float4*)&Ssc[q][kk][12];
            float m = sv[0];
            #pragma unroll
            for (int h2 = 1; h2 < 16; ++h2) m = fmaxf(m, sv[h2]);
            float sum = 0.f;
            #pragma unroll
            for (int h2 = 0; h2 < 16; ++h2) { sv[h2] = __expf(sv[h2] - m); sum += sv[h2]; }
            const float inv = 1.0f / sum;
            #pragma unroll
            for (int h2 = 0; h2 < 16; ++h2) Pm[h2][q][kk] = f2bf(sv[h2] * inv);
        }
        __syncthreads();

        // ---- phase 3: out += P * V, this wave's 16-wide d-chunk ----
        #pragma unroll
        for (int h = 0; h < HEADS; ++h) {
            BF16x16 Ap, Bv;
            Ap.u4[0] = *(const uint4*)&Pm[h][row][koff];
            Ap.u4[1] = *(const uint4*)&Pm[h][row][koff + 16];
            const unsigned short* vp =
                Vt + (((size_t)b*HEADS + h)*HDIM + wave*16 + n)*SEQ + kbase + koff;
            Bv.u4[0] = *(const uint4*)(vp + 0);
            Bv.u4[1] = *(const uint4*)(vp + 16);
            outAcc[h] = wmma_bf16(Ap, Bv, outAcc[h]);
        }
        __syncthreads();

        // hint next K tile into cache
        if (kb + 1 < SEQ/32) {
            const unsigned short* nk =
                Kp + (((size_t)b*HEADS + wave*4)*SEQ + kbase + 32 + n)*HDIM;
            __builtin_prefetch(nk, 0, 1);
        }
    }

    // epilogue: context in [token][emb] bf16 for the FC stage
    #pragma unroll
    for (int h = 0; h < HEADS; ++h) {
        const int e = h*HDIM + wave*16 + n;
        #pragma unroll
        for (int r = 0; r < 8; ++r) {
            const int s = q0 + qb + r;
            ctx[((size_t)b*SEQ + s)*EMB + e] = f2bf(outAcc[h][r]);
        }
    }
}

// ---------------------------------------------------------------------------
// Stage 3: out = ctx @ W_fc^T + b_fc   (4096 x 1024 x 1024), W pre-converted bf16
// ---------------------------------------------------------------------------
__global__ void __launch_bounds__(256) fc_kernel(
    const unsigned short* __restrict__ ctx, const unsigned short* __restrict__ Wb,
    const float* __restrict__ bias, float* __restrict__ out)
{
    const int wave = threadIdx.x >> 5;
    const int lane = threadIdx.x & 31;
    const int g  = blockIdx.x * 8 + wave;     // global 16x16 tile id
    const int tt = g >> 6;                    // token tile 0..255
    const int nt = g & 63;                    // out-emb tile 0..63
    const int t0 = tt*16, n0 = nt*16;

    const int row   = lane & 15;
    const int n     = lane & 15;
    const int koff  = (lane < 16) ? 0 : 8;
    const int bhalf = (lane < 16) ? 0 : 16;
    const int qb    = (lane < 16) ? 0 : 8;

    v8f acc = {0.f,0.f,0.f,0.f,0.f,0.f,0.f,0.f};
    for (int c = 0; c < EMB/32; ++c) {
        BF16x16 A, B;
        const unsigned short* ap = ctx + ((size_t)(t0 + row))*EMB + c*32 + koff;
        A.u4[0] = *(const uint4*)(ap);
        A.u4[1] = *(const uint4*)(ap + 16);
        const unsigned short* wp = Wb + (size_t)(n0 + n)*EMB + c*32 + koff;
        B.u4[0] = *(const uint4*)(wp);
        B.u4[1] = *(const uint4*)(wp + 16);
        acc = wmma_bf16(A, B, acc);
    }
    const float bv = bias[n0 + n];
    #pragma unroll
    for (int r = 0; r < 8; ++r)
        out[((size_t)(t0 + qb + r))*EMB + n0 + n] = acc[r] + bv;
}

// ---------------------------------------------------------------------------
extern "C" void kernel_launch(void* const* d_in, const int* in_sizes, int n_in,
                              void* d_out, int out_size, void* d_ws, size_t ws_size,
                              hipStream_t stream) {
    (void)in_sizes; (void)n_in; (void)out_size; (void)ws_size;
    const float* q   = (const float*)d_in[0];
    const float* k   = (const float*)d_in[1];
    const float* v   = (const float*)d_in[2];
    const float* Wq  = (const float*)d_in[3];
    const float* Wk  = (const float*)d_in[4];
    const float* Wv  = (const float*)d_in[5];
    const float* Wfc = (const float*)d_in[6];
    const float* bfc = (const float*)d_in[7];
    float* out = (float*)d_out;

    const size_t perTensor = (size_t)BATCH*HEADS*SEQ*HDIM;   // 4M bf16 elements
    unsigned short* Qp  = (unsigned short*)d_ws;
    unsigned short* Kp  = Qp + perTensor;
    unsigned short* Vt  = Kp + perTensor;
    unsigned short* ctx = Vt + perTensor;
    unsigned short* Wb  = ctx + (size_t)BATCH*SEQ*EMB;       // bf16 copy of W_fc

    cvt_kernel <<<(EMB*EMB/8 + 255)/256, 256, 0, stream>>>(Wfc, Wb, EMB*EMB/8);
    proj_kernel<<<BATCH*HEADS*8, 128, 0, stream>>>(q, Wq, Qp, 0);
    proj_kernel<<<BATCH*HEADS*8, 128, 0, stream>>>(k, Wk, Kp, 0);
    proj_kernel<<<BATCH*HEADS*8, 128, 0, stream>>>(v, Wv, Vt, 1);
    attn_kernel<<<BATCH*(SEQ/16), 128, 0, stream>>>(Qp, Kp, Vt, ctx);
    fc_kernel<<<(BATCH*SEQ/16)*(EMB/16)/8, 256, 0, stream>>>(ctx, Wb, bfc, out);
}
